// timescale_ouroboros_35562329211548
// MI455X (gfx1250) — matrix operand order repaired and checked by reading, not compiled, not run on hardware
//
#include <hip/hip_runtime.h>
#include <stdint.h>
#include <stddef.h>

// Problem constants (from reference)
#define BB   4
#define LL   1024
#define TOK  (BB * LL)      // 4096 tokens

// ---------------- CDNA5 WMMA types ----------------
typedef __bf16    v16bf __attribute__((ext_vector_type(16)));
typedef float     v8f   __attribute__((ext_vector_type(8)));
typedef uint32_t  u32x4 __attribute__((ext_vector_type(4)));

union FragBF { u32x4 q[2]; v16bf v; };

__device__ __forceinline__ uint16_t f2bf(float f) {
    uint32_t u = __float_as_uint(f);
    u += 0x7fffu + ((u >> 16) & 1u);      // round-to-nearest-even
    return (uint16_t)(u >> 16);
}
__device__ __forceinline__ float sil(float v) { return v / (1.f + __expf(-v)); }

// ---------------- Weight pack: (N,K) f32 row-major -> B-fragment-native bf16 ----------------
// Layout per 16(N)x32(K) tile: 256 u32 words; word index = lane*8 + j,
// lane%16 = N-within-tile, lane/16 = K-half h; word j packs K = 8h+2j+8*(j/4), +1 (ISA table order).
__global__ void pack_weight_bf16(const float* __restrict__ W, uint32_t* __restrict__ Wp,
                                 int N, int K) {
    int idx = blockIdx.x * blockDim.x + threadIdx.x;
    int ktiles = K >> 5;
    int total = (N >> 4) * ktiles * 256;
    if (idx >= total) return;
    int j    = idx & 7;
    int lane = (idx >> 3) & 31;
    int tile = idx >> 8;
    int nt = tile / ktiles, kt = tile - nt * ktiles;
    int k = (kt << 5) + ((lane >> 4) << 3) + (j << 1) + ((j >> 2) << 3);
    int n = (nt << 4) + (lane & 15);
    uint32_t lo = f2bf(W[(size_t)n * K + k]);
    uint32_t hi = f2bf(W[(size_t)n * K + k + 1]);
    Wp[idx] = lo | (hi << 16);
}

// ---------------- GEMM: C(M,N) = A(M,K) * W^T (+bias) (+=C)  via v_wmma_f32_16x16x32_bf16 ----
// A: row-major bf16. One 16x16 C tile per wave; 8 waves per block.
__global__ void __launch_bounds__(256) gemm_bf16_wmma(
    const uint16_t* __restrict__ A, const uint32_t* __restrict__ Wp,
    const float* __restrict__ bias, float* __restrict__ C,
    int M, int N, int K, int accumulate)
{
    const int lane   = threadIdx.x & 31;
    const int wave   = (blockIdx.x << 3) + (threadIdx.x >> 5);
    const int ntiles = N >> 4, mtiles = M >> 4, ktiles = K >> 5;
    if (wave >= mtiles * ntiles) return;           // whole-wave exit: EXEC stays all-ones
    const int mt = wave / ntiles, nt = wave - mt * ntiles;
    const int kh  = lane >> 4;                     // K-half select (ISA A-frag layout)
    const int row = (mt << 4) + (lane & 15);

    const uint16_t* arow = A  + (size_t)row * K + (kh << 3);
    const uint32_t* wp   = Wp + (((size_t)nt * ktiles) << 8) + (lane << 3);

    v8f acc = {};
    for (int kt = 0; kt < ktiles; ++kt) {
        FragBF a, b;
        a.q[0] = *(const u32x4*)(arow);            // K = kb + kh*8 .. +7
        a.q[1] = *(const u32x4*)(arow + 16);       // K = kb + 16 + kh*8 .. +7
        b.q[0] = *(const u32x4*)(wp);
        b.q[1] = *(const u32x4*)(wp + 4);
        __builtin_prefetch(arow + 32, 0, 3);       // global_prefetch_b8, next k-tile
        __builtin_prefetch(wp + 256, 0, 3);
        acc = __builtin_amdgcn_wmma_f32_16x16x32_bf16(
                  false, a.v, false, b.v, (short)0, acc, false, false);
        arow += 32;
        wp   += 256;
    }
    const int col = (nt << 4) + (lane & 15);       // C/D layout: lanes stripe N, VGPRs stripe M
    const int m0  = (mt << 4) + (kh << 3);
    const float bv = bias ? bias[col] : 0.f;
    #pragma unroll
    for (int i = 0; i < 8; ++i) {
        size_t o = (size_t)(m0 + i) * N + col;
        float v = acc[i] + bv;
        if (accumulate) v += C[o];
        C[o] = v;
    }
}

// ---------------- RMSNorm (one token per block) -> bf16 A-operand ----------------
__global__ void __launch_bounds__(256) rmsnorm_to_bf16(
    const float* __restrict__ U, const float* __restrict__ w,
    uint16_t* __restrict__ out, int dm)
{
    __shared__ float red[8];
    __shared__ float s_inv;
    const int t = blockIdx.x;
    const float* u = U + (size_t)t * dm;
    float s = 0.f;
    for (int i = threadIdx.x; i < dm; i += 256) { float v = u[i]; s += v * v; }
    for (int o = 16; o > 0; o >>= 1) s += __shfl_xor(s, o);
    if ((threadIdx.x & 31) == 0) red[threadIdx.x >> 5] = s;
    __syncthreads();
    if (threadIdx.x == 0) {
        float tot = 0.f;
        for (int i = 0; i < 8; ++i) tot += red[i];
        s_inv = rsqrtf(tot / (float)dm + 1e-5f);
    }
    __syncthreads();
    float inv = s_inv;
    for (int i = threadIdx.x; i < dm; i += 256)
        out[(size_t)t * dm + i] = f2bf(u[i] * inv * w[i]);
}

// ---------------- causal depthwise conv (K=4) + SiLU; emits f32 + bf16 ----------------
__global__ void conv_silu(const float* __restrict__ xz, const float* __restrict__ cw,
                          const float* __restrict__ cb, float* __restrict__ xi,
                          uint16_t* __restrict__ xibf, int dm)
{
    int idx = blockIdx.x * blockDim.x + threadIdx.x;
    if (idx >= TOK * dm) return;
    int e = idx % dm, tok = idx / dm;
    int t = tok & (LL - 1);
    const float* base = xz + (size_t)tok * (2 * dm) + e;   // xz[..., :dm] column e
    float v = cb[e];
    #pragma unroll
    for (int k = 0; k < 4; ++k) {
        int tt = t - 3 + k;
        if (tt >= 0) {
            long off = (long)(tt - t) * (long)(2 * dm);
            v += cw[e * 4 + k] * base[off];
        }
    }
    v = sil(v);
    xi[idx]   = v;
    xibf[idx] = f2bf(v);
}

// ---------------- delta = softplus(dbc[:, :R] @ dtW^T + dtb) ----------------
__global__ void dt_softplus(const float* __restrict__ dbc, const float* __restrict__ dtw,
                            const float* __restrict__ dtb, float* __restrict__ delta,
                            int dm, int R)
{
    int idx = blockIdx.x * blockDim.x + threadIdx.x;
    if (idx >= TOK * dm) return;
    int e = idx % dm, tok = idx / dm;
    const float* dr = dbc + (size_t)tok * (R + 32);
    const float* wr = dtw + (size_t)e * R;
    float s = dtb[e];
    for (int r = 0; r < R; ++r) s += dr[r] * wr[r];
    delta[idx] = (s > 20.f) ? s : log1pf(__expf(s));
}

// ---------------- SSM scan: h = exp(d*A)h + d*x*B ; y = <h,C> + D*x ----------------
// lane = (e_half<<4) | n ; 2 channels per wave, xor-shuffle reduction over N=16.
__global__ void __launch_bounds__(256) ssm_scan(
    const float* __restrict__ delta, const float* __restrict__ xi,
    const float* __restrict__ dbc, const float* __restrict__ A_log,
    const float* __restrict__ Dp, float* __restrict__ yout, int dm, int R)
{
    const int lane = threadIdx.x & 31;
    const int n = lane & 15;
    const int egroups = dm >> 4;
    const int b  = blockIdx.x / egroups;
    const int eg = blockIdx.x - b * egroups;
    const int e  = (eg << 4) + ((threadIdx.x >> 5) << 1) + (lane >> 4);
    const int rowlen = R + 32;
    const float a  = -__expf(A_log[e * 16 + n]);
    const float Dv = Dp[e];
    float h = 0.f;
    size_t tok = (size_t)b * LL;
    for (int t = 0; t < LL; ++t, ++tok) {
        float dl = delta[tok * dm + e];
        float xv = xi[tok * dm + e];
        const float* dr = dbc + tok * rowlen + R;
        h = __expf(dl * a) * h + dl * xv * dr[n];
        float s = h * dr[16 + n];
        s += __shfl_xor(s, 1); s += __shfl_xor(s, 2);
        s += __shfl_xor(s, 4); s += __shfl_xor(s, 8);
        if (n == 0) yout[tok * dm + e] = s + Dv * xv;
        __builtin_prefetch(dbc + (tok + 1) * rowlen, 0, 0);
    }
}

// ---------------- gate: yg = y_ssm * silu(z) -> bf16 ----------------
__global__ void gate_silu_bf16(const float* __restrict__ yssm, const float* __restrict__ xz,
                               uint16_t* __restrict__ out, int dm)
{
    int idx = blockIdx.x * blockDim.x + threadIdx.x;
    if (idx >= TOK * dm) return;
    int e = idx % dm, tok = idx / dm;
    float z = xz[(size_t)tok * (2 * dm) + dm + e];
    out[idx] = f2bf(yssm[idx] * sil(z));
}

// ---------------- misc elementwise ----------------
__global__ void cvt_bf16(const float* __restrict__ in, uint16_t* __restrict__ out, int n) {
    int idx = blockIdx.x * blockDim.x + threadIdx.x;
    if (idx < n) out[idx] = f2bf(in[idx]);
}
__global__ void build_uc(const float* __restrict__ x, const float* __restrict__ y,
                         float* __restrict__ uc) {
    int idx = blockIdx.x * blockDim.x + threadIdx.x;
    if (idx >= TOK * 128) return;
    int d = idx % 128, tok = idx / 128;
    float xv = x[idx], yv = y[idx];
    uc[(size_t)tok * 256 + d]       = yv - xv;   // dy
    uc[(size_t)tok * 256 + 128 + d] = yv;
}
__global__ void state_ic_k(const float* __restrict__ x, const float* __restrict__ y,
                           const float* __restrict__ iw, const float* __restrict__ ib,
                           float* __restrict__ sic) {
    int idx = blockIdx.x * blockDim.x + threadIdx.x;
    if (idx >= BB * 384) return;
    int j = idx % 384, b = idx / 384;
    const float* x0 = x + (size_t)b * LL * 128;
    const float* y0 = y + (size_t)b * LL * 128;
    const float* wr = iw + (size_t)j * 256;
    float s = ib[j];
    for (int d = 0; d < 128; ++d) s += wr[d]       * (y0[d] - x0[d]);
    for (int d = 0; d < 128; ++d) s += wr[128 + d] * y0[d];
    sic[idx] = sil(s);
}
__global__ void cumsum_state(const float* __restrict__ sp, const float* __restrict__ sic,
                             float* __restrict__ out_sp, float* __restrict__ ud) {
    int idx = blockIdx.x * blockDim.x + threadIdx.x;
    if (idx >= BB * 384) return;
    int j = idx % 384, b = idx / 384;
    float acc = sic[idx];
    size_t tok = (size_t)b * LL;
    for (int t = 0; t < LL; ++t, ++tok) {
        acc += sp[tok * 384 + j] * (1.0f / 50.0f);
        out_sp[tok * 384 + j] = acc;          // state_pred output
        ud[tok * 512 + j]     = acc;          // data-mamba input cols [0,384)
    }
}
__global__ void build_ud_x(const float* __restrict__ x, float* __restrict__ ud) {
    int idx = blockIdx.x * blockDim.x + threadIdx.x;
    if (idx >= TOK * 128) return;
    int d = idx % 128, tok = idx / 128;
    ud[(size_t)tok * 512 + 384 + d] = x[idx]; // data-mamba input cols [384,512)
}

// ---------------- host orchestration ----------------
struct LayerP { const float* p[10]; }; // norm_w,in_proj,conv_w,conv_b,x_proj,dt_W,dt_b,A_log,D,out_proj

static inline int cdiv(int a, int b) { return (a + b - 1) / b; }

extern "C" void kernel_launch(void* const* d_in, const int* in_sizes, int n_in,
                              void* d_out, int out_size, void* d_ws, size_t ws_size,
                              hipStream_t stream)
{
    (void)in_sizes; (void)n_in; (void)out_size; (void)ws_size;
    const float* x = (const float*)d_in[0];
    const float* y = (const float*)d_in[1];
    LayerP cl[2], dl[2];
    for (int l = 0; l < 2; ++l)
        for (int k = 0; k < 10; ++k) cl[l].p[k] = (const float*)d_in[2 + l * 10 + k];
    for (int l = 0; l < 2; ++l)
        for (int k = 0; k < 10; ++k) dl[l].p[k] = (const float*)d_in[22 + l * 10 + k];
    const float* cw = (const float*)d_in[42];
    const float* cb = (const float*)d_in[43];
    const float* iw = (const float*)d_in[44];
    const float* ib = (const float*)d_in[45];

    char* w = (char*)d_ws;
    auto grab = [&](size_t bytes) -> char* {
        char* p = w; w += (bytes + 255) & ~(size_t)255; return p;
    };
    float*    u_c   = (float*)   grab((size_t)TOK * 256 * 4);
    uint16_t* rn_bf = (uint16_t*)grab((size_t)TOK * 512 * 2);
    float*    xz    = (float*)   grab((size_t)TOK * 1024 * 4);
    float*    xi    = (float*)   grab((size_t)TOK * 512 * 4);
    uint16_t* xi_bf = (uint16_t*)grab((size_t)TOK * 512 * 2);
    float*    dbc   = (float*)   grab((size_t)TOK * 64 * 4);
    float*    dlt   = (float*)   grab((size_t)TOK * 512 * 4);
    float*    yssm  = (float*)   grab((size_t)TOK * 512 * 4);
    uint16_t* yg_bf = (uint16_t*)grab((size_t)TOK * 512 * 2);
    uint16_t* uc_bf = (uint16_t*)grab((size_t)TOK * 256 * 2);
    float*    sp    = (float*)   grab((size_t)TOK * 384 * 4);
    float*    sic   = (float*)   grab((size_t)BB * 384 * 4);
    uint32_t* wpk   = (uint32_t*)grab((size_t)1024 * 512 * 2 + 256);

    float* out_yhat = (float*)d_out;               // (B,L,512)
    float* out_sp   = out_yhat + (size_t)TOK * 512; // (B,L,384)
    float* u_d      = out_yhat;                     // data-mamba residual stream lives in d_out

    auto pack = [&](const float* W, int N, int K) {
        int total = (N >> 4) * (K >> 5) * 256;
        pack_weight_bf16<<<cdiv(total, 256), 256, 0, stream>>>(W, wpk, N, K);
    };
    auto gemm = [&](const uint16_t* A, const float* bias, float* C,
                    int M, int N, int K, int acc) {
        int waves = (M >> 4) * (N >> 4);
        gemm_bf16_wmma<<<cdiv(waves, 8), 256, 0, stream>>>(A, wpk, bias, C, M, N, K, acc);
    };
    auto layer = [&](float* u, int dm, const LayerP& P) {
        const int R = (dm == 256) ? 16 : 32;       // ceil(dm/16)
        rmsnorm_to_bf16<<<TOK, 256, 0, stream>>>(u, P.p[0], rn_bf, dm);
        pack(P.p[1], 2 * dm, dm);
        gemm(rn_bf, nullptr, xz, TOK, 2 * dm, dm, 0);                  // in_proj
        conv_silu<<<cdiv(TOK * dm, 256), 256, 0, stream>>>(xz, P.p[2], P.p[3], xi, xi_bf, dm);
        pack(P.p[4], R + 32, dm);
        gemm(xi_bf, nullptr, dbc, TOK, R + 32, dm, 0);                 // x_proj
        dt_softplus<<<cdiv(TOK * dm, 256), 256, 0, stream>>>(dbc, P.p[5], P.p[6], dlt, dm, R);
        ssm_scan<<<BB * (dm >> 4), 256, 0, stream>>>(dlt, xi, dbc, P.p[7], P.p[8], yssm, dm, R);
        gate_silu_bf16<<<cdiv(TOK * dm, 256), 256, 0, stream>>>(yssm, xz, yg_bf, dm);
        pack(P.p[9], dm, dm);
        gemm(yg_bf, nullptr, u, TOK, dm, dm, 1);                        // out_proj + residual
    };

    // control stack
    build_uc<<<cdiv(TOK * 128, 256), 256, 0, stream>>>(x, y, u_c);
    layer(u_c, 256, cl[0]);
    layer(u_c, 256, cl[1]);

    // state_pred = silu(ic) + cumsum(u_c @ cw^T + cb)/50
    cvt_bf16<<<cdiv(TOK * 256, 256), 256, 0, stream>>>(u_c, uc_bf, TOK * 256);
    pack(cw, 384, 256);
    gemm(uc_bf, cb, sp, TOK, 384, 256, 0);
    state_ic_k<<<cdiv(BB * 384, 256), 256, 0, stream>>>(x, y, iw, ib, sic);
    cumsum_state<<<cdiv(BB * 384, 256), 256, 0, stream>>>(sp, sic, out_sp, u_d);
    build_ud_x<<<cdiv(TOK * 128, 256), 256, 0, stream>>>(x, u_d);

    // data stack (in place in d_out -> yhat)
    layer(u_d, 512, dl[0]);
    layer(u_d, 512, dl[1]);
}
